// SelfAttention_11630771438205
// MI455X (gfx1250) — compile-verified, matrix operand order checked by reading, not run on hardware
//
#include <hip/hip_runtime.h>
#include <hip/hip_bf16.h>

typedef __attribute__((ext_vector_type(16))) __bf16 v16bf;
typedef __attribute__((ext_vector_type(8)))  __bf16 v8bf;
typedef __attribute__((ext_vector_type(8)))  float  v8f;

constexpr int S_  = 2048;
constexpr int B_  = 4;
constexpr int E_  = 1024;
constexpr int H_  = 16;
constexpr int HD_ = 64;
constexpr int M_TOK = S_ * B_;                 // 8192 tokens
constexpr size_t QKV_PART = (size_t)B_ * H_ * S_ * HD_;  // elements per Q/K/V

// ---------------------------------------------------------------------------
// gfx1250 async global->LDS copy (CDNA5 ISA). Writes LDS directly,
// tracked by ASYNCcnt (not LOADcnt/DScnt). LDS operand is the wave-relative
// LDS byte address = low 32 bits of the flat shared pointer.
// ---------------------------------------------------------------------------
__device__ __forceinline__ void async_load_b128(const void* gptr, void* lptr) {
    unsigned lds            = (unsigned)(size_t)lptr;
    unsigned long long gadr = (unsigned long long)(size_t)gptr;
    asm volatile("global_load_async_to_lds_b128 %0, %1, off"
                 :: "v"(lds), "v"(gadr) : "memory");
}

template <int N>
__device__ __forceinline__ void wait_asynccnt() {
#if __has_builtin(__builtin_amdgcn_s_wait_asynccnt)
    __builtin_amdgcn_s_wait_asynccnt(N);
#else
    asm volatile("s_wait_asynccnt %0" :: "n"(N) : "memory");
#endif
}

// Build a 16-element bf16 WMMA fragment from two contiguous 8-element spans.
// Per CDNA5 ISA (CDNA5 ISA): 16-bit A/B frag, lane=(half,m): elems 0..7 hold
// K = 8*half + i, elems 8..15 hold K = 16 + 8*half + i  -> two 16B loads.
__device__ __forceinline__ v16bf make_frag(const __bf16* lo_p, const __bf16* hi_p) {
    v8bf lo = *(const v8bf*)lo_p;
    v8bf hi = *(const v8bf*)hi_p;
    v16bf f;
#pragma unroll
    for (int i = 0; i < 8; ++i) { f[i] = lo[i]; f[i + 8] = hi[i]; }
    return f;
}

// Cross-lane max over each 16-lane half, entirely in VALU (dpp8 for xor-1/2/4,
// v_permlane16_b32 for xor-8) -- avoids ds_bpermute + s_wait_dscnt round trips.
#define DPP8_SEL(s0,s1,s2,s3,s4,s5,s6,s7) \
    ((s0)|((s1)<<3)|((s2)<<6)|((s3)<<9)|((s4)<<12)|((s5)<<15)|((s6)<<18)|((s7)<<21))

__device__ __forceinline__ float halfmax16(float x) {
#if __has_builtin(__builtin_amdgcn_mov_dpp8) && __has_builtin(__builtin_amdgcn_permlane16)
    int xi;
    xi = __builtin_amdgcn_mov_dpp8(__builtin_bit_cast(int, x),
                                   DPP8_SEL(1,0,3,2,5,4,7,6));      // xor 1
    x  = fmaxf(x, __builtin_bit_cast(float, xi));
    xi = __builtin_amdgcn_mov_dpp8(__builtin_bit_cast(int, x),
                                   DPP8_SEL(2,3,0,1,6,7,4,5));      // xor 2
    x  = fmaxf(x, __builtin_bit_cast(float, xi));
    xi = __builtin_amdgcn_mov_dpp8(__builtin_bit_cast(int, x),
                                   DPP8_SEL(4,5,6,7,0,1,2,3));      // xor 4
    x  = fmaxf(x, __builtin_bit_cast(float, xi));
    int xs = __builtin_bit_cast(int, x);                            // xor 8
    xi = __builtin_amdgcn_permlane16(xs, xs, 0xFEDCBA98u, 0x76543210u, false, false);
    x  = fmaxf(x, __builtin_bit_cast(float, xi));
#else
#pragma unroll
    for (int d = 1; d < 16; d <<= 1) x = fmaxf(x, __shfl_xor(x, d, 32));
#endif
    return x;
}

// ---------------------------------------------------------------------------
// fp32 -> bf16 convert (bandwidth-trivial)
// ---------------------------------------------------------------------------
__global__ void f32_to_bf16_kernel(const float* __restrict__ in,
                                   __bf16* __restrict__ out, int n) {
    int i = blockIdx.x * blockDim.x + threadIdx.x;
    if (i < n) out[i] = (__bf16)in[i];
}

// ---------------------------------------------------------------------------
// NT GEMM: C[M,N] = A[M,K] * W[N,K]^T + bias
// Block tile 128x128, BK=32, 8 waves (2x4), wave tile 64x32 = 4x2 WMMA tiles.
// Double-buffered LDS fed by async global->LDS; next k-tile streams in under
// the WMMAs (ASYNCcnt completes in order, so <=4 releases the current tile).
// MODE 0: QKV epilogue -> bf16 scatter into Q/K/V [B,H,S,HD], Q scaled 1/8.
// MODE 1: fp32 row-major output + bias.
// ---------------------------------------------------------------------------
template <int MODE>
__global__ __launch_bounds__(256)
void gemm_nt_bf16(const __bf16* __restrict__ A, const __bf16* __restrict__ W,
                  const float* __restrict__ bias,
                  __bf16* __restrict__ qkv_out, float* __restrict__ c_out,
                  int M, int N, int K) {
    constexpr int BM = 128, BN = 128, BK = 32, BKP = BK + 8;  // padded stride
    __shared__ alignas(16) __bf16 sA[2][BM * BKP];
    __shared__ alignas(16) __bf16 sB[2][BN * BKP];

    const int tid  = threadIdx.x;
    const int w    = tid >> 5, lane = tid & 31;
    const int half = lane >> 4, l16 = lane & 15;
    const int wm   = w >> 2, wn = w & 3;          // 2 waves in M, 4 in N
    const int m0   = blockIdx.x * BM, n0 = blockIdx.y * BN;

    v8f acc[4][2] = {};

    auto issue = [&](int buf, int k0) {           // 4 async b128 per thread
#pragma unroll
        for (int u = 0; u < 2; ++u) {
            int idx = tid + u * 256;              // 0..511
            int row = idx >> 2, q = idx & 3;      // 4 x 16B per 64B row
            async_load_b128(&A[(size_t)(m0 + row) * K + k0 + q * 8],
                            &sA[buf][row * BKP + q * 8]);
            async_load_b128(&W[(size_t)(n0 + row) * K + k0 + q * 8],
                            &sB[buf][row * BKP + q * 8]);
        }
    };
    issue(0, 0);

    for (int k0 = 0, it = 0; k0 < K; k0 += BK, ++it) {
        const int cur = it & 1;
        if (k0 + BK < K) { issue(cur ^ 1, k0 + BK); wait_asynccnt<4>(); }
        else             { wait_asynccnt<0>(); }
        __syncthreads();                          // cur tile visible to all

        v16bf af[4], bf[2];
#pragma unroll
        for (int i = 0; i < 4; ++i) {
            int r = wm * 64 + i * 16 + l16;
            af[i] = make_frag(&sA[cur][r * BKP + 8 * half],
                              &sA[cur][r * BKP + 16 + 8 * half]);
        }
#pragma unroll
        for (int j = 0; j < 2; ++j) {
            int r = wn * 32 + j * 16 + l16;
            bf[j] = make_frag(&sB[cur][r * BKP + 8 * half],
                              &sB[cur][r * BKP + 16 + 8 * half]);
        }
#pragma unroll
        for (int i = 0; i < 4; ++i)
#pragma unroll
            for (int j = 0; j < 2; ++j)
                acc[i][j] = __builtin_amdgcn_wmma_f32_16x16x32_bf16(
                    false, af[i], false, bf[j], (short)0, acc[i][j], false, false);
        __syncthreads();                          // done reading cur
    }

    // Epilogue. C/D layout: lane col n = l16, row m = v + 8*half (v = VGPR 0..7).
#pragma unroll
    for (int i = 0; i < 4; ++i) {
#pragma unroll
        for (int j = 0; j < 2; ++j) {
            int n = n0 + wn * 32 + j * 16 + l16;
            float bv = bias[n];
#pragma unroll
            for (int v = 0; v < 8; ++v) {
                int m = m0 + wm * 64 + i * 16 + v + 8 * half;
                float val = acc[i][j][v] + bv;
                if (MODE == 0) {
                    int part = n >> 10;            // 0=Q 1=K 2=V
                    int nin  = n & 1023;
                    int h = nin >> 6, d = nin & 63;
                    if (part == 0) val *= 0.125f;  // 1/sqrt(64) folded into Q
                    int s = m >> 2, b = m & 3;     // token m = s*B + b, B=4
                    size_t dst = (size_t)part * QKV_PART +
                                 (((size_t)(b * H_ + h) * S_ + s) * HD_ + d);
                    qkv_out[dst] = (__bf16)val;
                } else {
                    c_out[(size_t)m * N + n] = val;
                }
            }
        }
    }
}

// ---------------------------------------------------------------------------
// Flash attention. Grid = (B*H) * (S/128). 8 waves/WG, each wave owns 16 query
// rows; streams K/V in Bc=64 chunks. K chunk staged by async global->LDS;
// V goes through VGPRs (in-register transpose). Row-sum comes free from a
// ones-column appended to V; row-max reduction is pure-VALU dpp8/permlane16.
// ---------------------------------------------------------------------------
__global__ __launch_bounds__(256)
void flash_attn(const __bf16* __restrict__ Q, const __bf16* __restrict__ Kb,
                const __bf16* __restrict__ V, __bf16* __restrict__ Oout) {
    constexpr int BC = 64, HDP = HD_ + 8, BCP = BC + 8;
    __shared__ alignas(16) __bf16 sK[BC * HDP];          // [t][d], row-major
    __shared__ alignas(16) __bf16 sVt[(HD_ + 16) * BCP]; // [d][t] + ones tile
    __shared__ alignas(16) __bf16 sP[8][16 * BCP];       // per-wave P relayout

    const int tid  = threadIdx.x;
    const int w    = tid >> 5, lane = tid & 31;
    const int half = lane >> 4, l16 = lane & 15;
    const int qblocks = S_ / 128;
    const int bh  = blockIdx.x / qblocks;
    const int q0  = (blockIdx.x % qblocks) * 128 + w * 16;
    const size_t base = (size_t)bh * S_ * HD_;

    // V-extension rows 64..79: col of ones (row 64), zeros elsewhere. Written
    // once; staging below never touches rows >= 64. First barrier covers it.
    for (int idx = tid; idx < 16 * BC; idx += 256) {
        int r = idx >> 6, c = idx & 63;
        sVt[(HD_ + r) * BCP + c] = (r == 0) ? (__bf16)1.0f : (__bf16)0.0f;
    }

    // Q A-fragments (16 rows x 32 d), loaded once; scale already folded in.
    const __bf16* qrow = Q + base + (size_t)(q0 + l16) * HD_;
    v16bf qf[2];
#pragma unroll
    for (int kt = 0; kt < 2; ++kt)
        qf[kt] = make_frag(qrow + kt * 32 + 8 * half,
                           qrow + kt * 32 + 16 + 8 * half);

    v8f oacc[5] = {};            // 4 d-tiles + 1 row-sum tile
    float mrow[8];
#pragma unroll
    for (int v = 0; v < 8; ++v) mrow[v] = -1e30f;

    for (int t0 = 0; t0 < S_; t0 += BC) {
        // Stage K chunk [64][64] async; V transposed [64][64] via VGPRs.
#pragma unroll
        for (int u = 0; u < 2; ++u) {
            int idx = tid + u * 256;              // 0..511
            int row = idx >> 3, q = idx & 7;      // 8 x 16B per 128B row
            async_load_b128(&Kb[base + (size_t)(t0 + row) * HD_ + q * 8],
                            &sK[row * HDP + q * 8]);
            v8bf vv = *(const v8bf*)(&V[base + (size_t)(t0 + row) * HD_ + q * 8]);
#pragma unroll
            for (int e = 0; e < 8; ++e) sVt[(q * 8 + e) * BCP + row] = vv[e];
        }
        wait_asynccnt<0>();
        __syncthreads();

        // scores[16 q-rows][64 t] = Q * K^T  (4 n-tiles x 2 k-steps)
        v8f c[4] = {};
#pragma unroll
        for (int nt = 0; nt < 4; ++nt) {
            int t = nt * 16 + l16;
#pragma unroll
            for (int kt = 0; kt < 2; ++kt) {
                v16bf kf = make_frag(&sK[t * HDP + kt * 32 + 8 * half],
                                     &sK[t * HDP + kt * 32 + 16 + 8 * half]);
                c[nt] = __builtin_amdgcn_wmma_f32_16x16x32_bf16(
                    false, qf[kt], false, kf, (short)0, c[nt], false, false);
            }
        }

        // Online softmax per accumulator row v (row m = v + 8*half).
#pragma unroll
        for (int v = 0; v < 8; ++v) {
            float mx = fmaxf(fmaxf(c[0][v], c[1][v]), fmaxf(c[2][v], c[3][v]));
            mx = halfmax16(mx);                       // VALU dpp reduction
            float mnew = fmaxf(mrow[v], mx);
            float corr = __expf(mrow[v] - mnew);
            mrow[v] = mnew;
#pragma unroll
            for (int nt = 0; nt < 5; ++nt) oacc[nt][v] *= corr;  // incl. l tile
            int r = v + 8 * half;
#pragma unroll
            for (int nt = 0; nt < 4; ++nt) {
                float p = __expf(c[nt][v] - mnew);
                sP[w][r * BCP + nt * 16 + l16] = (__bf16)p;
            }
        }

        // P @ V(+ones). sP is per-wave; same-wave LDS ops are in-order.
        v16bf pf[2];
#pragma unroll
        for (int kt = 0; kt < 2; ++kt)
            pf[kt] = make_frag(&sP[w][l16 * BCP + kt * 32 + 8 * half],
                               &sP[w][l16 * BCP + kt * 32 + 16 + 8 * half]);
#pragma unroll
        for (int nt = 0; nt < 5; ++nt) {
            int d = nt * 16 + l16;
#pragma unroll
            for (int kt = 0; kt < 2; ++kt) {
                v16bf vf = make_frag(&sVt[d * BCP + kt * 32 + 8 * half],
                                     &sVt[d * BCP + kt * 32 + 16 + 8 * half]);
                oacc[nt] = __builtin_amdgcn_wmma_f32_16x16x32_bf16(
                    false, pf[kt], false, vf, (short)0, oacc[nt], false, false);
            }
        }
        __syncthreads();
    }

    // l lives in column 0 of the ones tile (lane half*16). Broadcast once.
    const int b = bh / H_, h = bh % H_;
#pragma unroll
    for (int v = 0; v < 8; ++v) {
        float lsum = __shfl(oacc[4][v], half * 16, 32);
        float inv  = 1.0f / lsum;
        int srow = q0 + v + 8 * half;
        size_t rowbase = ((size_t)srow * B_ + b) * E_ + (size_t)h * HD_;
#pragma unroll
        for (int nt = 0; nt < 4; ++nt)
            Oout[rowbase + nt * 16 + l16] = (__bf16)(oacc[nt][v] * inv);
    }
}

// ---------------------------------------------------------------------------
extern "C" void kernel_launch(void* const* d_in, const int* in_sizes, int n_in,
                              void* d_out, int out_size, void* d_ws, size_t ws_size,
                              hipStream_t stream) {
    const float* x     = (const float*)d_in[0];   // [S,B,E]
    const float* w_in  = (const float*)d_in[1];   // [3E,E]
    const float* b_in  = (const float*)d_in[2];   // [3E]
    const float* w_out = (const float*)d_in[3];   // [E,E]
    const float* b_out = (const float*)d_in[4];   // [E]
    float* out = (float*)d_out;                   // [S,B,E] fp32

    char* ws = (char*)d_ws;
    size_t off = 0;
    auto alloc = [&](size_t bytes) -> void* {
        void* p = ws + off;
        off += (bytes + 255) & ~(size_t)255;
        return p;
    };
    __bf16* xb  = (__bf16*)alloc((size_t)M_TOK * E_ * 2);      // 16 MB
    __bf16* wib = (__bf16*)alloc((size_t)3 * E_ * E_ * 2);     //  6 MB
    __bf16* wob = (__bf16*)alloc((size_t)E_ * E_ * 2);         //  2 MB
    __bf16* qkv = (__bf16*)alloc(3 * QKV_PART * 2);            // 48 MB
    __bf16* ao  = (__bf16*)alloc((size_t)M_TOK * E_ * 2);      // 16 MB
    (void)ws_size; (void)in_sizes; (void)n_in; (void)out_size;

    const int n1 = M_TOK * E_;        // x elements
    const int n2 = 3 * E_ * E_;       // w_in elements
    const int n3 = E_ * E_;           // w_out elements
    f32_to_bf16_kernel<<<(n1 + 255) / 256, 256, 0, stream>>>(x, xb, n1);
    f32_to_bf16_kernel<<<(n2 + 255) / 256, 256, 0, stream>>>(w_in, wib, n2);
    f32_to_bf16_kernel<<<(n3 + 255) / 256, 256, 0, stream>>>(w_out, wob, n3);

    // QKV projection: [8192,1024] x [3072,1024]^T -> scattered Q/K/V bf16
    gemm_nt_bf16<0><<<dim3(M_TOK / 128, (3 * E_) / 128), 256, 0, stream>>>(
        xb, wib, b_in, qkv, nullptr, M_TOK, 3 * E_, E_);

    // Flash attention over 64 (b,h) pairs x 16 query blocks
    flash_attn<<<B_ * H_ * (S_ / 128), 256, 0, stream>>>(
        qkv, qkv + QKV_PART, qkv + 2 * QKV_PART, ao);

    // Output projection: [8192,1024] x [1024,1024]^T -> fp32 out
    gemm_nt_bf16<1><<<dim3(M_TOK / 128, E_ / 128), 256, 0, stream>>>(
        ao, wob, b_out, nullptr, out, M_TOK, E_, E_);
}
